// DGCNN_29274497090206
// MI455X (gfx1250) — matrix operand order repaired and checked by reading, not compiled, not run on hardware
//
#include <hip/hip_runtime.h>

#define NPTS   1024
#define KNN_K  20
#define BATCH  8
#define NBLK   (NPTS / 16)
#define CCH    512            // xcat channels, point-major rows [b][n][512]

typedef __attribute__((ext_vector_type(16))) __bf16 v16bf;
typedef __attribute__((ext_vector_type(8)))  float  v8f;

// ---------------------------------------------------------------------------
// D = A(16x32) x B(32x16) + C, bf16 in / f32 acc. Per-lane operand layout
// (wave32) as loaded in this file, contiguous k for A and B:
//   A: lanes 0-15 -> M=lane,  K 0..15;         lanes 16-31 -> M=lane-16, K 16..31
//   B: lanes 0-15 -> N=lane,  K 0..15;         lanes 16-31 -> N=lane-16, K 16..31 (this code's order, not the ISA table order)
//   C/D: lanes 0-15 -> N=lane, vgpr r -> M=r;  lanes 16-31 -> N=lane-16, vgpr r -> M=r+8
// ---------------------------------------------------------------------------
__device__ __forceinline__ v8f wmma_bf16(v16bf a, v16bf b, v8f c)
{
  return __builtin_amdgcn_wmma_f32_16x16x32_bf16(false, a, false, b,
                                                 (short)0, c, false, false);
}

// Per-lane A operand for chunk cc of W[o0..o0+15][0..CTOT) (zero-padded).
template<int CTOT>
__device__ __forceinline__ v16bf load_A(const float* __restrict__ W, int o0,
                                        int lane, int cc)
{
  const int row  = lane & 15;
  const int koff = (lane >> 4) * 16;
  const float* wr = W + (o0 + row) * CTOT;
  v16bf a;
#pragma unroll
  for (int i = 0; i < 16; ++i) {
    const int c  = koff + i;   // {koff..+15}
    const int ch = cc * 32 + c;
    a[i] = (__bf16)((ch < CTOT) ? wr[ch] : 0.f);
  }
  return a;
}

// Stage all A chunks into LDS (per-lane operand order: each lane owns its slot).
template<int CTOT, int NCH>
__device__ __forceinline__ void stage_A(const float* __restrict__ W, int o0,
                                        int lane, __bf16* As)
{
#pragma unroll
  for (int cc = 0; cc < NCH; ++cc) {
    v16bf a = load_A<CTOT>(W, o0, lane, cc);
    *reinterpret_cast<v16bf*>(As + (cc * 32 + lane) * 16) = a;
  }
}

__device__ __forceinline__ v16bf fetch_A(const __bf16* As, int lane, int cc)
{
  return *reinterpret_cast<const v16bf*>(As + (cc * 32 + lane) * 16);
}

// Per-lane edge-feature B operand, chunk cc: f = [x[:,j]-x[:,n] ; x[:,n]].
// PST>0: point-major rows of stride PST (contiguous channels -> float4 loads).
// PST==0: channel-major with stride NPTS (layer 1 only, CIN=3).
template<int CIN, int PST>
__device__ __forceinline__ v16bf load_B_edge(const float* __restrict__ xb,
                                             int n, int j, int kb, int cc)
{
  constexpr int CTOT = 2 * CIN;
  v16bf b;
  if (PST > 0 && cc * 32 + 32 <= CIN) {
    const float4* pj = reinterpret_cast<const float4*>(xb + (long)j * PST + cc * 32 + kb);
    const float4* pn = reinterpret_cast<const float4*>(xb + (long)n * PST + cc * 32 + kb);
#pragma unroll
    for (int q = 0; q < 4; ++q) {
      const float4 a4 = pj[q], c4 = pn[q];
      b[4 * q + 0] = (__bf16)(a4.x - c4.x);
      b[4 * q + 1] = (__bf16)(a4.y - c4.y);
      b[4 * q + 2] = (__bf16)(a4.z - c4.z);
      b[4 * q + 3] = (__bf16)(a4.w - c4.w);
    }
  } else if (PST > 0) {
    const float4* pn = reinterpret_cast<const float4*>(xb + (long)n * PST + cc * 32 + kb - CIN);
#pragma unroll
    for (int q = 0; q < 4; ++q) {
      const float4 c4 = pn[q];
      b[4 * q + 0] = (__bf16)c4.x;
      b[4 * q + 1] = (__bf16)c4.y;
      b[4 * q + 2] = (__bf16)c4.z;
      b[4 * q + 3] = (__bf16)c4.w;
    }
  } else {
    // channel-major, mixed / zero-padded chunk (layer 1) -- branchless
#pragma unroll
    for (int i = 0; i < 16; ++i) {
      const int ch  = cc * 32 + kb + i;
      const int chn = (ch < CIN) ? ch : ((ch < CTOT) ? ch - CIN : 0);
      const float ctrv = xb[chn * NPTS + n];
      const float xv   = xb[chn * NPTS + j];
      b[i] = (__bf16)((ch < CIN) ? (xv - ctrv) : ((ch < CTOT) ? ctrv : 0.f));
    }
  }
  return b;
}

// Per-lane B operand from a point-major (N, 512) feature map.
__device__ __forceinline__ v16bf load_B_flat(const float* __restrict__ xb,
                                             int n, int kb, int cc)
{
  const float4* p = reinterpret_cast<const float4*>(xb + (long)n * CCH + cc * 32 + kb);
  v16bf b;
#pragma unroll
  for (int q = 0; q < 4; ++q) {
    const float4 c4 = p[q];
    b[4 * q + 0] = (__bf16)c4.x;
    b[4 * q + 1] = (__bf16)c4.y;
    b[4 * q + 2] = (__bf16)c4.z;
    b[4 * q + 3] = (__bf16)c4.w;
  }
  return b;
}

// ---------------------------------------------------------------------------
// kNN: one wave per (b, n) row; generic element strides (cs per channel, ms per
// point). neg_dist(n,m) = -||x_n - x_m||^2; K rounds of wave-wide argmax with
// jax-style tie-break (lower index wins).
// ---------------------------------------------------------------------------
__global__ void __launch_bounds__(32, 1)
knn_kernel(const float* __restrict__ xin, int bstride,
           int cs, int ms, int C, int* __restrict__ idx_out)
{
  const int row  = blockIdx.x;                 // 0 .. B*N-1
  const int b    = row / NPTS, n = row % NPTS;
  const int lane = threadIdx.x;                // wave32
  const float* xb = xin + (long)b * bstride;

  __shared__ float xn[128];
  for (int c = lane; c < C; c += 32) xn[c] = xb[c * cs + n * ms];
  __syncthreads();

  float d[32];
#pragma unroll 4
  for (int j = 0; j < 32; ++j) {
    const int m = lane + 32 * j;
    const float* xm = xb + (long)m * ms;
    float acc = 0.f;
    for (int c = 0; c < C; ++c) {
      float t = xn[c] - xm[c * cs];
      acc += t * t;
    }
    d[j] = -acc;
  }

  for (int r = 0; r < KNN_K; ++r) {
    float best = -3.402823e38f; int bj = 0;
#pragma unroll
    for (int j = 0; j < 32; ++j)
      if (d[j] > best) { best = d[j]; bj = j; }
    int bm = lane + 32 * bj;
#pragma unroll
    for (int off = 16; off > 0; off >>= 1) {
      float ov = __shfl_xor(best, off, 32);
      int   om = __shfl_xor(bm,   off, 32);
      if (ov > best || (ov == best && om < bm)) { best = ov; bm = om; }
    }
    if (lane == 0) idx_out[row * KNN_K + r] = bm;
#pragma unroll
    for (int j = 0; j < 32; ++j)
      if (((bm & 31) == lane) && ((bm >> 5) == j)) d[j] = -3.402823e38f;
  }
}

// ---------------------------------------------------------------------------
// EdgeConv pass 1: per-channel sum / sumsq of y over (b, n, k).
// One wave owns its 16 output channels -> deterministic, no atomics.
// ---------------------------------------------------------------------------
template<int CIN, int PST>
__global__ void __launch_bounds__(32, 1)
edgeconv_stats_k(const float* __restrict__ xin, int bstride,
                 const int* __restrict__ idx,
                 const float* __restrict__ W,
                 float* __restrict__ ssum, float* __restrict__ ssq)
{
  constexpr int CTOT = 2 * CIN;
  constexpr int NCH  = (CTOT + 31) / 32;
  const int o0   = blockIdx.x * 16;
  const int lane = threadIdx.x;
  const int nl   = lane & 15;
  const int kb   = (lane >> 4) * 16;

  __shared__ alignas(32) __bf16 As[NCH * 512];
  stage_A<CTOT, NCH>(W, o0, lane, As);
  __syncthreads();

  float sum8[8], sq8[8];
#pragma unroll
  for (int r = 0; r < 8; ++r) { sum8[r] = 0.f; sq8[r] = 0.f; }

  for (int b = 0; b < BATCH; ++b) {
    const float* xb = xin + (long)b * bstride;
    const int*   ib = idx + b * NPTS * KNN_K;
    for (int n = nl; n < NPTS; n += 16) {
      for (int k = 0; k < KNN_K; ++k) {
        const int j = ib[n * KNN_K + k];
        v8f acc = {};
#pragma unroll
        for (int cc = 0; cc < NCH; ++cc)
          acc = wmma_bf16(fetch_A(As, lane, cc),
                          load_B_edge<CIN, PST>(xb, n, j, kb, cc), acc);
#pragma unroll
        for (int r = 0; r < 8; ++r) { float v = acc[r]; sum8[r] += v; sq8[r] += v * v; }
      }
    }
  }
#pragma unroll
  for (int r = 0; r < 8; ++r) {
    float s = sum8[r], q = sq8[r];
    for (int off = 1; off < 16; off <<= 1) {
      s += __shfl_xor(s, off, 32);
      q += __shfl_xor(q, off, 32);
    }
    if (lane == 0)       { ssum[o0 + r]     = s; ssq[o0 + r]     = q; }
    else if (lane == 16) { ssum[o0 + 8 + r] = s; ssq[o0 + 8 + r] = q; }
  }
}

// ---------------------------------------------------------------------------
// EdgeConv pass 2: recompute tiles, BN affine + leaky-relu, max over k.
// Writes point-major output rows: xout[b][n][o0+half*8 .. +8) as 2x float4.
// grid = (BATCH*NBLK, O/16), one wave per block.
// ---------------------------------------------------------------------------
template<int CIN, int PST>
__global__ void __launch_bounds__(32, 1)
edgeconv_apply_k(const float* __restrict__ xin, int bstride,
                 const int* __restrict__ idx,
                 const float* __restrict__ W,
                 const float* __restrict__ scl,
                 const float* __restrict__ shf,
                 float* __restrict__ xout, int out_bstride)
{
  constexpr int CTOT = 2 * CIN;
  constexpr int NCH  = (CTOT + 31) / 32;
  const int b    = blockIdx.x >> 6;                 // NBLK == 64
  const int n0   = (blockIdx.x & 63) * 16;
  const int o0   = blockIdx.y * 16;
  const int lane = threadIdx.x;
  const int nl   = lane & 15;
  const int kb   = (lane >> 4) * 16;
  const int half = lane >> 4;
  const int n    = n0 + nl;

  const float* xb = xin + (long)b * bstride;
  const int*   ib = idx + b * NPTS * KNN_K;

  __shared__ alignas(32) __bf16 As[NCH * 512];
  stage_A<CTOT, NCH>(W, o0, lane, As);
  __syncthreads();

  float sc[8], sh[8], rmax[8];
#pragma unroll
  for (int r = 0; r < 8; ++r) {
    const int o = o0 + r + half * 8;
    sc[r] = scl[o]; sh[r] = shf[o]; rmax[r] = -3.402823e38f;
  }

  for (int k = 0; k < KNN_K; ++k) {
    const int j = ib[n * KNN_K + k];
    v8f acc = {};
#pragma unroll
    for (int cc = 0; cc < NCH; ++cc)
      acc = wmma_bf16(fetch_A(As, lane, cc),
                      load_B_edge<CIN, PST>(xb, n, j, kb, cc), acc);
#pragma unroll
    for (int r = 0; r < 8; ++r) {
      float y = acc[r] * sc[r] + sh[r];
      y = (y >= 0.f) ? y : 0.2f * y;
      rmax[r] = fmaxf(rmax[r], y);
    }
  }

  float* orow = xout + (long)b * out_bstride + (long)n * CCH + o0 + half * 8;
  reinterpret_cast<float4*>(orow)[0] = make_float4(rmax[0], rmax[1], rmax[2], rmax[3]);
  reinterpret_cast<float4*>(orow)[1] = make_float4(rmax[4], rmax[5], rmax[6], rmax[7]);
}

// ---------------------------------------------------------------------------
// Final 512 -> 1024 pointwise conv, pass 1 (per-channel stats, deterministic).
// ---------------------------------------------------------------------------
__global__ void __launch_bounds__(32, 1)
mlp_stats(const float* __restrict__ xcat,
          const float* __restrict__ W,
          float* __restrict__ ssum, float* __restrict__ ssq)
{
  const int o0   = blockIdx.x * 16;
  const int lane = threadIdx.x;
  const int nl   = lane & 15;
  const int kb   = (lane >> 4) * 16;

  __shared__ alignas(32) __bf16 As[16 * 512];
  stage_A<512, 16>(W, o0, lane, As);
  __syncthreads();

  float sum8[8], sq8[8];
#pragma unroll
  for (int r = 0; r < 8; ++r) { sum8[r] = 0.f; sq8[r] = 0.f; }

  for (int b = 0; b < BATCH; ++b) {
    const float* xb = xcat + (long)b * CCH * NPTS;
    for (int n = nl; n < NPTS; n += 16) {
      v8f acc = {};
#pragma unroll
      for (int cc = 0; cc < 16; ++cc)
        acc = wmma_bf16(fetch_A(As, lane, cc), load_B_flat(xb, n, kb, cc), acc);
#pragma unroll
      for (int r = 0; r < 8; ++r) { float v = acc[r]; sum8[r] += v; sq8[r] += v * v; }
    }
  }
#pragma unroll
  for (int r = 0; r < 8; ++r) {
    float s = sum8[r], q = sq8[r];
    for (int off = 1; off < 16; off <<= 1) {
      s += __shfl_xor(s, off, 32);
      q += __shfl_xor(q, off, 32);
    }
    if (lane == 0)       { ssum[o0 + r]     = s; ssq[o0 + r]     = q; }
    else if (lane == 16) { ssum[o0 + 8 + r] = s; ssq[o0 + 8 + r] = q; }
  }
}

// ---------------------------------------------------------------------------
// Final conv pass 2: BN affine + lrelu + global max over n -> out[b, o].
// grid = (BATCH, 1024/16), one wave per block.
// ---------------------------------------------------------------------------
__global__ void __launch_bounds__(32, 1)
mlp_apply(const float* __restrict__ xcat,
          const float* __restrict__ W,
          const float* __restrict__ scl,
          const float* __restrict__ shf,
          float* __restrict__ out)
{
  const int b    = blockIdx.x;
  const int o0   = blockIdx.y * 16;
  const int lane = threadIdx.x;
  const int nl   = lane & 15;
  const int kb   = (lane >> 4) * 16;
  const int half = lane >> 4;

  __shared__ alignas(32) __bf16 As[16 * 512];
  stage_A<512, 16>(W, o0, lane, As);
  __syncthreads();

  float sc[8], sh[8], rmax[8];
#pragma unroll
  for (int r = 0; r < 8; ++r) {
    const int o = o0 + r + half * 8;
    sc[r] = scl[o]; sh[r] = shf[o]; rmax[r] = -3.402823e38f;
  }

  const float* xb = xcat + (long)b * CCH * NPTS;
  for (int n = nl; n < NPTS; n += 16) {
    v8f acc = {};
#pragma unroll
    for (int cc = 0; cc < 16; ++cc)
      acc = wmma_bf16(fetch_A(As, lane, cc), load_B_flat(xb, n, kb, cc), acc);
#pragma unroll
    for (int r = 0; r < 8; ++r) {
      float y = acc[r] * sc[r] + sh[r];
      y = (y >= 0.f) ? y : 0.2f * y;
      rmax[r] = fmaxf(rmax[r], y);
    }
  }
#pragma unroll
  for (int r = 0; r < 8; ++r) {
    float m = rmax[r];
    for (int off = 1; off < 16; off <<= 1)
      m = fmaxf(m, __shfl_xor(m, off, 32));
    if (lane == 0)       out[b * 1024 + o0 + r]     = m;
    else if (lane == 16) out[b * 1024 + o0 + 8 + r] = m;
  }
}

// ---------------------------------------------------------------------------
// BN finalize: scale = gamma * rsqrt(var + eps); shift = beta - mean * scale.
// ---------------------------------------------------------------------------
__global__ void __launch_bounds__(256)
finalize_stats(const float* __restrict__ ssum,
               const float* __restrict__ ssq,
               const float* __restrict__ gamma,
               const float* __restrict__ beta,
               float* __restrict__ scl, float* __restrict__ shf,
               int O, float invcnt)
{
  int o = blockIdx.x * 256 + threadIdx.x;
  if (o >= O) return;
  float m = ssum[o] * invcnt;
  float v = ssq[o] * invcnt - m * m;
  float a = gamma[o] * rsqrtf(v + 1e-5f);
  scl[o] = a;
  shf[o] = beta[o] - m * a;
}

// ---------------------------------------------------------------------------
extern "C" void kernel_launch(void* const* d_in, const int* in_sizes, int n_in,
                              void* d_out, int out_size, void* d_ws, size_t ws_size,
                              hipStream_t stream)
{
  const float* x  = (const float*)d_in[0];
  const float* W1 = (const float*)d_in[1];
  const float* g1 = (const float*)d_in[2];
  const float* b1 = (const float*)d_in[3];
  const float* W2 = (const float*)d_in[4];
  const float* g2 = (const float*)d_in[5];
  const float* b2 = (const float*)d_in[6];
  const float* W3 = (const float*)d_in[7];
  const float* g3 = (const float*)d_in[8];
  const float* b3 = (const float*)d_in[9];
  const float* W4 = (const float*)d_in[10];
  const float* g4 = (const float*)d_in[11];
  const float* b4 = (const float*)d_in[12];
  const float* W5 = (const float*)d_in[13];
  const float* g5 = (const float*)d_in[14];
  const float* b5 = (const float*)d_in[15];
  float* out = (float*)d_out;

  // workspace: xcat point-major (8, 1024, 512) f32 | idx (8,1024,20) i32 | stats
  float* xcat = (float*)d_ws;
  int*   idxb = (int*)(xcat + (size_t)BATCH * NPTS * CCH);
  float* ssum = (float*)(idxb + (size_t)BATCH * NPTS * KNN_K);
  float* ssq  = ssum + 1024;
  float* scl  = ssq + 1024;
  float* shf  = scl + 1024;

  const float inv2d = 1.f / (float)(BATCH * NPTS * KNN_K);
  const float inv1d = 1.f / (float)(BATCH * NPTS);
  const int   CATBS = CCH * NPTS;      // xcat per-batch stride (elements)

  // ---- layer 1: x(3ch, channel-major) -> x1(64ch) at xcat[..][0:64]
  knn_kernel<<<BATCH * NPTS, 32, 0, stream>>>(x, 3 * NPTS, NPTS, 1, 3, idxb);
  edgeconv_stats_k<3, 0><<<64 / 16, 32, 0, stream>>>(x, 3 * NPTS, idxb, W1, ssum, ssq);
  finalize_stats<<<1, 256, 0, stream>>>(ssum, ssq, g1, b1, scl, shf, 64, inv2d);
  edgeconv_apply_k<3, 0><<<dim3(BATCH * NBLK, 64 / 16), 32, 0, stream>>>(
      x, 3 * NPTS, idxb, W1, scl, shf, xcat + 0, CATBS);

  // ---- layer 2: x1(64ch) -> x2(64ch) at xcat[..][64:128]
  knn_kernel<<<BATCH * NPTS, 32, 0, stream>>>(xcat + 0, CATBS, 1, CCH, 64, idxb);
  edgeconv_stats_k<64, CCH><<<64 / 16, 32, 0, stream>>>(xcat + 0, CATBS, idxb, W2,
                                                        ssum, ssq);
  finalize_stats<<<1, 256, 0, stream>>>(ssum, ssq, g2, b2, scl, shf, 64, inv2d);
  edgeconv_apply_k<64, CCH><<<dim3(BATCH * NBLK, 64 / 16), 32, 0, stream>>>(
      xcat + 0, CATBS, idxb, W2, scl, shf, xcat + 64, CATBS);

  // ---- layer 3: x2(64ch) -> x3(128ch) at xcat[..][128:256]
  knn_kernel<<<BATCH * NPTS, 32, 0, stream>>>(xcat + 64, CATBS, 1, CCH, 64, idxb);
  edgeconv_stats_k<64, CCH><<<128 / 16, 32, 0, stream>>>(xcat + 64, CATBS, idxb, W3,
                                                         ssum, ssq);
  finalize_stats<<<1, 256, 0, stream>>>(ssum, ssq, g3, b3, scl, shf, 128, inv2d);
  edgeconv_apply_k<64, CCH><<<dim3(BATCH * NBLK, 128 / 16), 32, 0, stream>>>(
      xcat + 64, CATBS, idxb, W3, scl, shf, xcat + 128, CATBS);

  // ---- layer 4: x3(128ch) -> x4(256ch) at xcat[..][256:512]
  knn_kernel<<<BATCH * NPTS, 32, 0, stream>>>(xcat + 128, CATBS, 1, CCH, 128, idxb);
  edgeconv_stats_k<128, CCH><<<256 / 16, 32, 0, stream>>>(xcat + 128, CATBS, idxb, W4,
                                                          ssum, ssq);
  finalize_stats<<<1, 256, 0, stream>>>(ssum, ssq, g4, b4, scl, shf, 256, inv2d);
  edgeconv_apply_k<128, CCH><<<dim3(BATCH * NBLK, 256 / 16), 32, 0, stream>>>(
      xcat + 128, CATBS, idxb, W4, scl, shf, xcat + 256, CATBS);

  // ---- final 512 -> 1024 conv + BN + lrelu + max over n
  mlp_stats<<<1024 / 16, 32, 0, stream>>>(xcat, W5, ssum, ssq);
  finalize_stats<<<4, 256, 0, stream>>>(ssum, ssq, g5, b5, scl, shf, 1024, inv1d);
  mlp_apply<<<dim3(BATCH, 1024 / 16), 32, 0, stream>>>(xcat, W5, scl, shf, out);
}
